// PrismaticSpectralAttention_19705309954382
// MI455X (gfx1250) — matrix operand order repaired and checked by reading, not compiled, not run on hardware
//
#include <hip/hip_runtime.h>
#include <hip/hip_bf16.h>

typedef __attribute__((ext_vector_type(16))) __bf16 v16bf;
typedef __attribute__((ext_vector_type(8)))  __bf16 v8bf;
typedef __attribute__((ext_vector_type(8)))  float  v8f;

#define T_    8
#define NS_   196
#define D_    768
#define H_    12
#define G_    4
#define HD_   64
#define S_    1568      // T_*NS_
#define NTOK_ 1569
#define NPAD_ 1584      // 99*16
#define QKVN_ 2304
#define SCALE_ 0.125f
#define PI2_  6.28318530717958647692f

// ---------------- helpers ----------------

static __device__ __forceinline__ v8f zero8() {
  v8f z;
#pragma unroll
  for (int i = 0; i < 8; ++i) z[i] = 0.f;
  return z;
}

static __device__ __forceinline__ v8bf ld8bf(const __bf16* p) {
  return *reinterpret_cast<const v8bf*>(p);
}

static __device__ __forceinline__ v16bf cat8(v8bf lo, v8bf hi) {
  v16bf r;
#pragma unroll
  for (int i = 0; i < 8; ++i) { r[i] = lo[i]; r[i + 8] = hi[i]; }
  return r;
}

// A operand: 16x32 bf16 tile from row-major [M][K] at (row0,k0).
// ISA layout: lane l holds row (row0 + l&15); K elems {koff..koff+7, 16+koff..16+koff+7}, koff = (l>=16)*8.
static __device__ __forceinline__ v16bf mkA(const __bf16* base, int ld, int row0, int k0, int lane) {
  int row  = row0 + (lane & 15);
  int koff = (lane >> 4) * 8;
  const __bf16* p = base + (long)row * ld + k0;
  return cat8(ld8bf(p + koff), ld8bf(p + 16 + koff));
}

// B operand: 32x16 bf16, B(k,n) = Bt[n][k] where Bt is row-major [N][K].
// ISA layout: lane l holds column (n0 + l&15); K elems {koff..koff+7, 16+koff..16+koff+7}, koff = (l>=16)*8.
static __device__ __forceinline__ v16bf mkB(const __bf16* bt, int ld, int n0, int k0, int lane) {
  int col = n0 + (lane & 15);
  int kk  = k0 + (lane >> 4) * 8;
  const __bf16* p = bt + (long)col * ld + kk;
  return cat8(ld8bf(p), ld8bf(p + 16));
}

static __device__ __forceinline__ v8f wmma_bf16(v16bf a, v16bf b, v8f c) {
  return __builtin_amdgcn_wmma_f32_16x16x32_bf16(false, a, false, b, (short)0, c, false, false);
}

// ---------------- kernel 0: f32 -> bf16 weight conversion (once per launch; weights reused ~100x) --------

__global__ void cvtw_kernel(const float* __restrict__ src, __bf16* __restrict__ dst, long n) {
  long i = (long)blockIdx.x * blockDim.x + threadIdx.x;
  if (i < n) dst[i] = (__bf16)src[i];
}

// ---------------- kernel 1: band-filtered features (FFT band filter = 8x8 linear mix over T) ----------------

__global__ void bandfeat_kernel(const float* __restrict__ x, __bf16* __restrict__ bfeat) {
  long idx = (long)blockIdx.x * blockDim.x + threadIdx.x;
  const long total = (long)G_ * S_ * D_;
  if (idx >= total) return;
  int d  = (int)(idx % D_);
  long sd = idx / D_;
  int s  = (int)(sd % S_);
  int b  = (int)(sd / S_);
  int t  = s / NS_;
  int ns = s % NS_;
  int fs = b * 2;
  int fe = (b == G_ - 1) ? T_ : fs + 2;
  if (fs > 5) fs = 5;
  if (fe > 5) fe = 5;
  float acc = 0.f;
  if (fs < 5) {
#pragma unroll
    for (int tp = 0; tp < T_; ++tp) {
      float m = 0.f;
      for (int f = fs; f < fe; ++f) {
        float w = (f == 0 || f == 4) ? 1.f : 2.f;
        m += w * __cosf(PI2_ * (float)f * (float)(t - tp) * 0.125f);
      }
      m *= 0.125f;
      acc += m * x[(long)(1 + tp * NS_ + ns) * D_ + d];
    }
  }
  bfeat[idx] = (__bf16)acc;
}

// ---------------- kernel 2: QKV GEMM (bf16 WMMA), route into Q, K, V^T ----------------

__global__ __launch_bounds__(128) void qkv_kernel(const __bf16* __restrict__ bfeat,
                                                  const __bf16* __restrict__ wq,     // bf16 [G][2304][768]
                                                  const float* __restrict__ qkv_b,
                                                  __bf16* __restrict__ qb,
                                                  __bf16* __restrict__ kb,
                                                  __bf16* __restrict__ vtb) {
  int lane = threadIdx.x & 31;
  int wave = threadIdx.x >> 5;
  int m0 = blockIdx.x * 16;                // 0..1552
  int n0 = blockIdx.y * 64 + wave * 16;    // 0..2288
  int b  = blockIdx.z;
  const __bf16* A = bfeat + (long)b * S_ * D_;
  const __bf16* W = wq + (long)b * QKVN_ * D_;
  v8f c = zero8();
  for (int k0 = 0; k0 < D_; k0 += 32) {
    if (k0 + 32 < D_) {   // prefetch next k-tile (global_prefetch_b8)
      __builtin_prefetch(A + (long)(m0 + (lane & 15)) * D_ + k0 + 32, 0, 1);
      __builtin_prefetch(W + (long)(n0 + (lane & 15)) * D_ + k0 + 32, 0, 1);
    }
    v16bf a  = mkA(A, D_, m0, k0, lane);
    v16bf bm = mkB(W, D_, n0, k0, lane);
    c = wmma_bf16(a, bm, c);
  }
  int col   = n0 + (lane & 15);
  float bias = qkv_b[b * QKVN_ + col];
  int rbase = m0 + 8 * (lane >> 4);
#pragma unroll
  for (int r = 0; r < 8; ++r) {
    int row = rbase + r;
    float v = c[r] + bias;
    if (col < 768) {
      int h = col >> 6, d = col & 63;
      qb[(((long)b * H_ + h) * S_ + row) * HD_ + d] = (__bf16)v;
    } else if (col < 1536) {
      int j = col - 768; int h = j >> 6, d = j & 63;
      kb[(((long)b * H_ + h) * S_ + row) * HD_ + d] = (__bf16)v;
    } else {
      int j = col - 1536; int h = j >> 6, d = j & 63;
      vtb[(((long)b * H_ + h) * HD_ + d) * S_ + row] = (__bf16)v;  // V transposed: [64][S]
    }
  }
}

// ---------------- kernel 3: flash attention, one wave per (band, head, 16-query block) ----------------

__global__ __launch_bounds__(256) void attn_kernel(const __bf16* __restrict__ qb,
                                                   const __bf16* __restrict__ kb,
                                                   const __bf16* __restrict__ vtb,
                                                   const float* __restrict__ spec_filters,
                                                   __bf16* __restrict__ bandout) {
  __shared__ __align__(16) __bf16 pLds[8][16][32];   // per-wave P staging tile
  int lane = threadIdx.x & 31;
  int wave = threadIdx.x >> 5;
  int wid  = blockIdx.x * 8 + wave;   // 0..4703 = 48*98
  int qt = wid % 98;
  int bh = wid / 98;                  // band*H + head, 0..47
  int h  = bh % H_;
  int b  = bh / H_;
  const __bf16* Q  = qb  + (long)bh * S_ * HD_;
  const __bf16* K  = kb  + (long)bh * S_ * HD_;
  const __bf16* VT = vtb + (long)bh * HD_ * S_;
  float sc = SCALE_ * spec_filters[b * H_ + h];
  int m0 = qt * 16;

  // Q block as A operands (K-dim 0..31 and 32..63), loaded once
  v16bf qa0 = mkA(Q, HD_, m0, 0, lane);
  v16bf qa1 = mkA(Q, HD_, m0, 32, lane);

  v8f acc[4];
#pragma unroll
  for (int i = 0; i < 4; ++i) acc[i] = zero8();
  float m_run[8], l_run[8];
#pragma unroll
  for (int r = 0; r < 8; ++r) { m_run[r] = -3.0e38f; l_run[r] = 0.f; }

  int rr = 8 * (lane >> 4);
  int cc = lane & 15;

  for (int kt = 0; kt < S_ / 32; ++kt) {
    int key0 = kt * 32;
    if (kt + 1 < S_ / 32) {  // prefetch next key tile of K rows
      __builtin_prefetch(K + (long)(key0 + 32 + (lane & 31)) * HD_, 0, 1);
    }
    // scores for 32 keys: two 16x16 C tiles
    v8f s0 = zero8(), s1 = zero8();
    s0 = wmma_bf16(qa0, mkB(K, HD_, key0, 0, lane), s0);
    s0 = wmma_bf16(qa1, mkB(K, HD_, key0, 32, lane), s0);
    s1 = wmma_bf16(qa0, mkB(K, HD_, key0 + 16, 0, lane), s1);
    s1 = wmma_bf16(qa1, mkB(K, HD_, key0 + 16, 32, lane), s1);

    // online softmax update (rows live across 16-lane halves)
#pragma unroll
    for (int r = 0; r < 8; ++r) {
      float a0 = s0[r] * sc;
      float a1 = s1[r] * sc;
      float t = fmaxf(a0, a1);
#pragma unroll
      for (int msk = 1; msk < 16; msk <<= 1) t = fmaxf(t, __shfl_xor(t, msk, 16));
      float mnew = fmaxf(m_run[r], t);
      float corr = __expf(m_run[r] - mnew);
      float p0 = __expf(a0 - mnew);
      float p1 = __expf(a1 - mnew);
      float rs = p0 + p1;
#pragma unroll
      for (int msk = 1; msk < 16; msk <<= 1) rs += __shfl_xor(rs, msk, 16);
      l_run[r] = l_run[r] * corr + rs;
      m_run[r] = mnew;
#pragma unroll
      for (int cch = 0; cch < 4; ++cch) acc[cch][r] *= corr;
      s0[r] = p0;
      s1[r] = p1;
    }

    // re-layout P (C layout -> A layout) through LDS
#pragma unroll
    for (int r = 0; r < 8; ++r) {
      pLds[wave][rr + r][cc]      = (__bf16)s0[r];
      pLds[wave][rr + r][16 + cc] = (__bf16)s1[r];
    }
    __syncthreads();
    v16bf pa = mkA(&pLds[wave][0][0], 32, 0, 0, lane);
    __syncthreads();

    // P @ V : B(k=key, n=dim) = VT[dim][key0+k] -> contiguous rows of VT
#pragma unroll
    for (int cch = 0; cch < 4; ++cch) {
      acc[cch] = wmma_bf16(pa, mkB(VT, S_, cch * 16, key0, lane), acc[cch]);
    }
  }

  // normalize and store
#pragma unroll
  for (int cch = 0; cch < 4; ++cch) {
#pragma unroll
    for (int r = 0; r < 8; ++r) {
      int row = m0 + rr + r;
      float v = acc[cch][r] / l_run[r];
      bandout[((long)b * S_ + row) * D_ + h * 64 + cch * 16 + cc] = (__bf16)v;
    }
  }
}

// ---------------- kernel 4: fusion scores (per-band block reduction) ----------------

__global__ __launch_bounds__(256) void fscore_kernel(const __bf16* __restrict__ bandout,
                                                     const float* __restrict__ fusion_w,
                                                     const float* __restrict__ fusion_b,
                                                     float* __restrict__ scores) {
  __shared__ float red[256];
  int b = blockIdx.x;
  const __bf16* p = bandout + (long)b * S_ * D_;
  float acc = 0.f;
  for (long i = threadIdx.x; i < (long)S_ * D_; i += 256) {
    int d = (int)(i % D_);
    acc += (float)p[i] * fusion_w[d];
  }
  red[threadIdx.x] = acc;
  __syncthreads();
  for (int st = 128; st > 0; st >>= 1) {
    if ((int)threadIdx.x < st) red[threadIdx.x] += red[threadIdx.x + st];
    __syncthreads();
  }
  if (threadIdx.x == 0) scores[b] = red[0] / (float)S_ + fusion_b[0];
}

__global__ void fsoftmax_kernel(const float* __restrict__ scores, float* __restrict__ wts) {
  if (threadIdx.x == 0) {
    float m = fmaxf(fmaxf(scores[0], scores[1]), fmaxf(scores[2], scores[3]));
    float e0 = __expf(scores[0] - m), e1 = __expf(scores[1] - m);
    float e2 = __expf(scores[2] - m), e3 = __expf(scores[3] - m);
    float inv = 1.f / (e0 + e1 + e2 + e3);
    wts[0] = e0 * inv; wts[1] = e1 * inv; wts[2] = e2 * inv; wts[3] = e3 * inv;
  }
}

// ---------------- kernel 5: weighted band fusion + CLS, padded to 1584 rows, bf16 ----------------

__global__ void fuse_kernel(const float* __restrict__ x,
                            const __bf16* __restrict__ bandout,
                            const float* __restrict__ wts,
                            __bf16* __restrict__ fused) {
  long idx = (long)blockIdx.x * blockDim.x + threadIdx.x;
  const long total = (long)NPAD_ * D_;
  if (idx >= total) return;
  int d   = (int)(idx % D_);
  int row = (int)(idx / D_);
  float v = 0.f;
  if (row == 0) {
    v = x[d];                                // CLS token
  } else if (row <= S_) {
    int s = row - 1;
#pragma unroll
    for (int b = 0; b < G_; ++b)
      v += wts[b] * (float)bandout[((long)b * S_ + s) * D_ + d];
  }
  fused[idx] = (__bf16)v;
}

// ---------------- kernel 6: output projection (bf16 WMMA) ----------------

__global__ __launch_bounds__(128) void outproj_kernel(const __bf16* __restrict__ fused,
                                                      const __bf16* __restrict__ wo,   // bf16 [768][768]
                                                      const float* __restrict__ out_b,
                                                      float* __restrict__ out) {
  int lane = threadIdx.x & 31;
  int wave = threadIdx.x >> 5;
  int m0 = blockIdx.x * 16;                // 0..1568 (padded rows are zero)
  int n0 = blockIdx.y * 64 + wave * 16;
  v8f c = zero8();
  for (int k0 = 0; k0 < D_; k0 += 32) {
    if (k0 + 32 < D_) {
      __builtin_prefetch(fused + (long)(m0 + (lane & 15)) * D_ + k0 + 32, 0, 1);
      __builtin_prefetch(wo + (long)(n0 + (lane & 15)) * D_ + k0 + 32, 0, 1);
    }
    v16bf a  = mkA(fused, D_, m0, k0, lane);
    v16bf bm = mkB(wo, D_, n0, k0, lane);
    c = wmma_bf16(a, bm, c);
  }
  int col = n0 + (lane & 15);
  float bias = out_b[col];
  int rbase = m0 + 8 * (lane >> 4);
#pragma unroll
  for (int r = 0; r < 8; ++r) {
    int row = rbase + r;
    if (row < NTOK_) out[(long)row * D_ + col] = c[r] + bias;
  }
}

// ---------------- launcher ----------------

extern "C" void kernel_launch(void* const* d_in, const int* in_sizes, int n_in,
                              void* d_out, int out_size, void* d_ws, size_t ws_size,
                              hipStream_t stream) {
  const float* x     = (const float*)d_in[0];
  const float* qkv_w = (const float*)d_in[1];
  const float* qkv_b = (const float*)d_in[2];
  const float* spec  = (const float*)d_in[3];
  const float* fus_w = (const float*)d_in[4];
  const float* fus_b = (const float*)d_in[5];
  const float* out_w = (const float*)d_in[6];
  const float* out_b = (const float*)d_in[7];
  float* out = (float*)d_out;

  char* ws = (char*)d_ws;
  size_t off = 0;
  auto take = [&](size_t bytes) -> char* {
    char* p = ws + off;
    off = (off + bytes + 255) & ~(size_t)255;
    return p;
  };
  const size_t bandBytes = (size_t)G_ * S_ * D_ * sizeof(__bf16);
  __bf16* bfeat = (__bf16*)take(bandBytes);
  __bf16* qbuf  = (__bf16*)take(bandBytes);
  __bf16* kbuf  = (__bf16*)take(bandBytes);
  __bf16* vtbuf = (__bf16*)take(bandBytes);
  __bf16* bout  = (__bf16*)take(bandBytes);
  __bf16* fused = (__bf16*)take((size_t)NPAD_ * D_ * sizeof(__bf16));
  __bf16* wqkv  = (__bf16*)take((size_t)G_ * QKVN_ * D_ * sizeof(__bf16));
  __bf16* wout  = (__bf16*)take((size_t)D_ * D_ * sizeof(__bf16));
  float*  scores = (float*)take(256);
  float*  wts    = (float*)take(256);
  (void)in_sizes; (void)n_in; (void)out_size; (void)ws_size;

  {
    long n = (long)G_ * QKVN_ * D_;
    cvtw_kernel<<<(int)((n + 255) / 256), 256, 0, stream>>>(qkv_w, wqkv, n);
  }
  {
    long n = (long)D_ * D_;
    cvtw_kernel<<<(int)((n + 255) / 256), 256, 0, stream>>>(out_w, wout, n);
  }
  {
    long total = (long)G_ * S_ * D_;
    int blocks = (int)((total + 255) / 256);
    bandfeat_kernel<<<blocks, 256, 0, stream>>>(x, bfeat);
  }
  {
    dim3 g(S_ / 16, QKVN_ / 64, G_);
    qkv_kernel<<<g, 128, 0, stream>>>(bfeat, wqkv, qkv_b, qbuf, kbuf, vtbuf);
  }
  attn_kernel<<<588, 256, 0, stream>>>(qbuf, kbuf, vtbuf, spec, bout);
  fscore_kernel<<<G_, 256, 0, stream>>>(bout, fus_w, fus_b, scores);
  fsoftmax_kernel<<<1, 32, 0, stream>>>(scores, wts);
  {
    long total = (long)NPAD_ * D_;
    int blocks = (int)((total + 255) / 256);
    fuse_kernel<<<blocks, 256, 0, stream>>>(x, bout, wts, fused);
  }
  {
    dim3 g(NPAD_ / 16, D_ / 64);
    outproj_kernel<<<g, 128, 0, stream>>>(fused, wout, out_b, out);
  }
}
